// LSTMModel_764504179521
// MI455X (gfx1250) — matrix operand order repaired and checked by reading, not compiled, not run on hardware
//
#include <hip/hip_runtime.h>

#define Bn  128
#define Tn  256
#define Tm1 255
#define Fn  6
#define Hn  512
#define G4  2048
#define Ln  4
#define Sn  32
#define Dn  3

typedef __attribute__((ext_vector_type(16))) __bf16 v16bf;
typedef __attribute__((ext_vector_type(8)))  float  v8f;
typedef __attribute__((ext_vector_type(4)))  float  v4f;
typedef __attribute__((ext_vector_type(4)))  unsigned int v4u;
typedef __attribute__((ext_vector_type(8)))  int v8i;
typedef __attribute__((ext_vector_type(4)))  int v4i;

union BFrag { v16bf v; v4f f[2]; };

__device__ __forceinline__ unsigned short f2bf(float x) {
  unsigned int u = __float_as_uint(x);
  u += 0x7FFFu + ((u >> 16) & 1u);
  return (unsigned short)(u >> 16);
}
__device__ __forceinline__ float bf2f(unsigned short b) {
  return __uint_as_float(((unsigned int)b) << 16);
}
__device__ __forceinline__ float sigmoidf_(float x) {
  return 1.f / (1.f + __expf(-x));
}
__device__ __forceinline__ float tanhf_(float x) {
  float e = __expf(2.f * x);
  return (e - 1.f) / (e + 1.f);
}

// ---- Tensor Data Mover: 2D tile (tile_k elems per row, tile_rows rows) of
// bf16 data, global (row_stride_elems pitch) -> LDS (packed), per ISA D# layout.
__device__ __forceinline__ void tdm_issue(v4u g0, v8i g1) {
  v4i z4 = {0, 0, 0, 0};
#if defined(__clang_major__) && (__clang_major__ >= 23)
  v8i z8 = {0, 0, 0, 0, 0, 0, 0, 0};
  __builtin_amdgcn_tensor_load_to_lds(g0, g1, z4, z4, z8, 0);
#else
  __builtin_amdgcn_tensor_load_to_lds(g0, g1, z4, z4, 0);
#endif
}
__device__ __forceinline__ void tdm_load_2d_bf16(const void* gptr, unsigned lds_off,
                                                 int tile_k, int tile_rows,
                                                 int row_stride_elems) {
  unsigned long long ga = (unsigned long long)gptr;
  v4u g0;
  g0[0] = 1u;                                              // count=1, user descriptor
  g0[1] = lds_off;                                         // lds_addr (bytes)
  g0[2] = (unsigned int)ga;                                // global_addr[31:0]
  g0[3] = ((unsigned int)(ga >> 32) & 0x01FFFFFFu) | (2u << 30);  // addr[56:32] | type=2
  v8i g1;
  g1[0] = (1 << 16);                                       // data_size = 2 bytes
  g1[1] = (tile_k & 0xFFFF) << 16;                         // tensor_dim0[15:0]
  g1[2] = ((tile_k >> 16) & 0xFFFF) | ((tile_rows & 0xFFFF) << 16);  // dim0 hi | tensor_dim1 lo
  g1[3] = ((tile_rows >> 16) & 0xFFFF) | ((tile_k & 0xFFFF) << 16);  // dim1 hi | tile_dim0
  g1[4] = (tile_rows & 0xFFFF);                            // tile_dim1 (tile_dim2=0)
  g1[5] = row_stride_elems;                                // tensor_dim0_stride[31:0]
  g1[6] = 0;
  g1[7] = 0;
  tdm_issue(g0, g1);
}

// A fragment (16-bit A 16x32, MxK). A points at tile row 0, row-major, lda elems.
__device__ __forceinline__ v16bf load_a_frag(const unsigned short* A, int lda, int k0) {
  int lane = threadIdx.x & 31;
  const unsigned short* p = A + (lane & 15) * lda + k0 + ((lane >> 4) << 3);
  BFrag r;
  r.f[0] = *(const v4f*)p;
  r.f[1] = *(const v4f*)(p + 16);
  return r.v;
}
// B fragment from an N-major [rows][K] panel (LDS or global pointer).
// lanes 0-15 hold K=k0..k0+7,k0+16..k0+23; lanes 16-31 hold K=k0+8..k0+15,k0+24..k0+31; N = lane%16.
__device__ __forceinline__ v16bf load_b_frag(const unsigned short* BT, int ldb, int k0) {
  int lane = threadIdx.x & 31;
  const unsigned short* p = BT + (lane & 15) * ldb + k0 + ((lane >> 4) << 3);
  BFrag r;
  r.f[0] = *(const v4f*)p;
  r.f[1] = *(const v4f*)(p + 16);
  return r.v;
}
__device__ __forceinline__ v8f wmma_bf16(v16bf a, v16bf b, v8f c) {
  return __builtin_amdgcn_wmma_f32_16x16x32_bf16(false, a, false, b, (short)0, c, false, false);
}

// --- transpose + convert U / Wk weights to bf16, N-major [G4][Hn] ---
__global__ void k_prep(const float* __restrict__ U, const float* __restrict__ Wk,
                       unsigned short* __restrict__ UT, unsigned short* __restrict__ WkT) {
  long long idx = (long long)blockIdx.x * blockDim.x + threadIdx.x;
  const long long total = 7LL * Hn * G4;
  if (idx >= total) return;
  int mat = (int)(idx / ((long long)Hn * G4));
  long long rem = idx % ((long long)Hn * G4);
  int k = (int)(rem / G4);
  int n = (int)(rem % G4);
  if (mat < 4) {
    float v = U[(long long)mat * Hn * G4 + (long long)k * G4 + n];
    UT[(long long)mat * G4 * Hn + (long long)n * Hn + k] = f2bf(v);
  } else {
    float v = Wk[(long long)(mat - 4) * Hn * G4 + (long long)k * G4 + n];
    WkT[(long long)(mat - 4) * G4 * Hn + (long long)n * Hn + k] = f2bf(v);
  }
}

// --- UniformDiff + layer0 input projection (K=6) + bias -> bf16 z-buffer ---
__global__ void k_z0(const float* __restrict__ x, const float* __restrict__ centre,
                     const float* __restrict__ bound, const float* __restrict__ W0,
                     const float* __restrict__ b0, unsigned short* __restrict__ z) {
  long long idx = (long long)blockIdx.x * blockDim.x + threadIdx.x;
  const long long total = (long long)Tm1 * Bn * G4;
  if (idx >= total) return;
  int n = (int)(idx % G4);
  long long tb = idx / G4;
  int b = (int)(tb % Bn);
  int t = (int)(tb / Bn);
  float acc = b0[n];
  #pragma unroll
  for (int f = 0; f < Fn; ++f) {
    float inv = 1.f / bound[f];
    float d = (x[((long long)b * Tn + t + 1) * Fn + f] -
               x[((long long)b * Tn + t) * Fn + f]) * inv;   // centre cancels in diff
    acc += d * W0[f * G4 + n];
  }
  z[idx] = f2bf(acc);
}

// --- batched input projection for layers 1..3: Z = Hseq @ W + b, bf16 WMMA ---
// grid (Tm1, G4/128), 256 threads. B panel (128 x 512 bf16 = 128 KB) staged to
// LDS once per block via TDM; inner loop ping-pongs the B fragment (2 live
// buffers) so the ds_load for tile j+1 issues before the WMMA for tile j,
// and pipelines the next A fragment across the WMMA chain.
__global__ void __launch_bounds__(256)
k_gemm_in(const unsigned short* __restrict__ Aseq, const unsigned short* __restrict__ BT,
          const float* __restrict__ bias, unsigned short* __restrict__ Z) {
  extern __shared__ unsigned short Bs[];   // [128][Hn]
  int wave = threadIdx.x >> 5;
  int lane = threadIdx.x & 31;
  int m0 = blockIdx.x * 128 + wave * 16;
  int nbase = blockIdx.y * 128;

  if (wave == 0) {  // async-stage whole B panel for this block
    tdm_load_2d_bf16(BT + (long long)nbase * Hn, 0u, Hn, 128, Hn);
  }

  v8f acc[8];
  #pragma unroll
  for (int j = 0; j < 8; ++j) {
    float bv = bias[nbase + j * 16 + (lane & 15)];
    #pragma unroll
    for (int r = 0; r < 8; ++r) acc[j][r] = bv;
  }

  const unsigned short* Abase = Aseq + (long long)m0 * Hn;
  v16bf a = load_a_frag(Abase, Hn, 0);     // overlaps with TDM copy

  if (wave == 0) __builtin_amdgcn_s_wait_tensorcnt(0);
  __syncthreads();

  for (int k0 = 0; k0 < Hn; k0 += 32) {
    v16bf bcur = load_b_frag(Bs, Hn, k0);
    v16bf anext = a;
    if (k0 + 32 < Hn) anext = load_a_frag(Abase, Hn, k0 + 32);
    #pragma unroll
    for (int j = 0; j < 8; ++j) {
      v16bf bnext = bcur;
      if (j < 7) bnext = load_b_frag(Bs + (long long)((j + 1) * 16) * Hn, Hn, k0);
      acc[j] = wmma_bf16(a, bcur, acc[j]);
      bcur = bnext;
    }
    a = anext;
  }

  int rowoff = (lane >> 4) << 3;
  #pragma unroll
  for (int j = 0; j < 8; ++j) {
    int n = nbase + j * 16 + (lane & 15);
    #pragma unroll
    for (int r = 0; r < 8; ++r) {
      int m = m0 + r + rowoff;
      Z[(long long)m * G4 + n] = f2bf(acc[j][r]);
    }
  }
}

// --- one recurrent LSTM step: z_t + h_{t-1} @ U, then gates ---
// grid = Hn/16 = 32 blocks, 256 threads. Wave w owns rows [16w,16w+16),
// block owns h-cols [16*bid,16*bid+16). The 4 gate U panels (16 x 512 each)
// are TDM-staged to LDS (64 KB) while the z_t accumulator init loads run.
__global__ void __launch_bounds__(256)
k_lstm_step(const unsigned short* __restrict__ Zt, const unsigned short* __restrict__ Hprev,
            const unsigned short* __restrict__ UT, float* __restrict__ Cst,
            unsigned short* __restrict__ Hout, float* __restrict__ HfinT, int first) {
  extern __shared__ unsigned short Us[];   // [4*16][Hn]
  int wave = threadIdx.x >> 5;
  int lane = threadIdx.x & 31;
  int m0 = wave * 16;
  int nh0 = blockIdx.x * 16;
  int rowoff = (lane >> 4) << 3;
  int ncol = nh0 + (lane & 15);

  if (!first && wave == 0) {
    #pragma unroll
    for (int g = 0; g < 4; ++g) {
      tdm_load_2d_bf16(UT + (long long)(g * Hn + nh0) * Hn,
                       (unsigned)(g * 16 * Hn * 2), Hn, 16, Hn);
    }
  }

  v8f acc[4];
  #pragma unroll
  for (int g = 0; g < 4; ++g) {
    #pragma unroll
    for (int r = 0; r < 8; ++r) {
      int m = m0 + r + rowoff;
      acc[g][r] = bf2f(Zt[(long long)m * G4 + g * Hn + ncol]);
    }
  }

  if (!first) {
    const unsigned short* Abase = Hprev + (long long)m0 * Hn;
    v16bf a = load_a_frag(Abase, Hn, 0);   // overlaps with TDM copy
    if (wave == 0) __builtin_amdgcn_s_wait_tensorcnt(0);
    __syncthreads();
    for (int k0 = 0; k0 < Hn; k0 += 32) {
      v16bf bcur = load_b_frag(Us, Hn, k0);
      v16bf anext = a;
      if (k0 + 32 < Hn) anext = load_a_frag(Abase, Hn, k0 + 32);
      #pragma unroll
      for (int g = 0; g < 4; ++g) {
        v16bf bnext = bcur;
        if (g < 3) bnext = load_b_frag(Us + (long long)((g + 1) * 16) * Hn, Hn, k0);
        acc[g] = wmma_bf16(a, bcur, acc[g]);
        bcur = bnext;
      }
      a = anext;
    }
  }

  #pragma unroll
  for (int r = 0; r < 8; ++r) {
    int m = m0 + r + rowoff;
    long long hi = (long long)m * Hn + ncol;
    float cp = first ? 0.f : Cst[hi];
    float ig = sigmoidf_(acc[0][r]);
    float fg = sigmoidf_(acc[1][r]);
    float gg = tanhf_(acc[2][r]);
    float og = sigmoidf_(acc[3][r]);
    float c = fg * cp + ig * gg;
    float h = og * tanhf_(c);
    Cst[hi] = c;
    Hout[hi] = f2bf(h);
    if (HfinT) HfinT[hi] = h;
  }
}

// --- final dense + Restore ---
__global__ void k_out(const float* __restrict__ Hfin, const float* __restrict__ Wd,
                      const float* __restrict__ bd, const float* __restrict__ centre,
                      const float* __restrict__ bound, float* __restrict__ out) {
  int idx = blockIdx.x * blockDim.x + threadIdx.x;
  if (idx >= Bn * Sn * Dn) return;
  int d = idx % Dn;
  int s = (idx / Dn) % Sn;
  int b = idx / (Dn * Sn);
  float acc = bd[d];
  const float* hp = Hfin + ((long long)s * Bn + b) * Hn;
  for (int k = 0; k < Hn; ++k) acc += hp[k] * Wd[k * Dn + d];
  out[idx] = acc;                                         // out
  out[Bn * Sn * Dn + idx] = acc * bound[d] + centre[d];   // pred
}

extern "C" void kernel_launch(void* const* d_in, const int* in_sizes, int n_in,
                              void* d_out, int out_size, void* d_ws, size_t ws_size,
                              hipStream_t stream) {
  const float* x      = (const float*)d_in[0];
  const float* centre = (const float*)d_in[1];
  const float* bound  = (const float*)d_in[2];
  const float* W0     = (const float*)d_in[3];
  const float* Wk     = (const float*)d_in[4];
  const float* U      = (const float*)d_in[5];
  const float* bias   = (const float*)d_in[6];
  const float* Wd     = (const float*)d_in[7];
  const float* bd     = (const float*)d_in[8];
  float* out = (float*)d_out;

  char* ws = (char*)d_ws;
  size_t off = 0;
  unsigned short* UT  = (unsigned short*)(ws + off); off += (size_t)Ln * G4 * Hn * 2;
  unsigned short* WkT = (unsigned short*)(ws + off); off += (size_t)3 * G4 * Hn * 2;
  unsigned short* Z   = (unsigned short*)(ws + off); off += (size_t)Tm1 * Bn * G4 * 2;
  unsigned short* HA  = (unsigned short*)(ws + off); off += (size_t)Tm1 * Bn * Hn * 2;
  unsigned short* HB  = (unsigned short*)(ws + off); off += (size_t)Tm1 * Bn * Hn * 2;
  float* Cst  = (float*)(ws + off); off += (size_t)Bn * Hn * 4;
  float* Hfin = (float*)(ws + off);

  {
    long long total = 7LL * Hn * G4;
    k_prep<<<(unsigned)((total + 255) / 256), 256, 0, stream>>>(U, Wk, UT, WkT);
  }
  {
    long long total = (long long)Tm1 * Bn * G4;
    k_z0<<<(unsigned)((total + 255) / 256), 256, 0, stream>>>(x, centre, bound, W0, bias, Z);
  }

  unsigned short* hseq[2] = {HA, HB};
  for (int l = 0; l < Ln; ++l) {
    unsigned short* hcur = hseq[l & 1];
    if (l > 0) {
      unsigned short* hin = hseq[(l - 1) & 1];
      dim3 grid(Tm1, G4 / 128);
      k_gemm_in<<<grid, 256, 128 * Hn * 2, stream>>>(hin, WkT + (size_t)(l - 1) * G4 * Hn,
                                                     bias + (size_t)l * G4, Z);
    }
    for (int t = 0; t < Tm1; ++t) {
      float* hf = (l == Ln - 1 && t >= Tm1 - Sn)
                      ? (Hfin + (size_t)(t - (Tm1 - Sn)) * Bn * Hn) : (float*)nullptr;
      k_lstm_step<<<Hn / 16, 256, 64 * Hn * 2, stream>>>(
          Z + (size_t)t * Bn * G4,
          (t > 0) ? (hcur + (size_t)(t - 1) * Bn * Hn) : hcur,
          UT + (size_t)l * G4 * Hn, Cst,
          hcur + (size_t)t * Bn * Hn, hf, t == 0 ? 1 : 0);
    }
  }
  k_out<<<(Bn * Sn * Dn + 255) / 256, 256, 0, stream>>>(Hfin, Wd, bd, centre, bound, out);
}
